// QuantumRegressionModel_65481071400898
// MI455X (gfx1250) — matrix operand order repaired and checked by reading, not compiled, not run on hardware
//
#include <hip/hip_runtime.h>
#include <hip/hip_bf16.h>
#include <stdint.h>

#define N_TOK 8192
#define DIM   256
#define JCHUNK 32

typedef __attribute__((ext_vector_type(16))) _Float16 v16h;
typedef __attribute__((ext_vector_type(8)))  _Float16 v8h;
typedef __attribute__((ext_vector_type(8)))  float    v8f;

static __device__ __forceinline__ v8f wmma16(v16h a, v16h b, v8f c) {
  // D(f32 16x16) = A(16x32 f16) * B(32x16 f16) + C
  return __builtin_amdgcn_wmma_f32_16x16x32_f16(false, a, false, b, (short)0, c, false, false);
}

static __device__ __forceinline__ v16h cat8(v8h lo, v8h hi) {
  return __builtin_shufflevector(lo, hi, 0,1,2,3,4,5,6,7,8,9,10,11,12,13,14,15);
}

static __device__ __forceinline__ v8f vzero8() {
  v8f z = {0.f,0.f,0.f,0.f,0.f,0.f,0.f,0.f};
  return z;
}

// A-fragment (16x32 f16) from row-major [*, DIM] source; lane: row = ln,
// elements 0..15 at k=c*32+hf*16+s.
static __device__ __forceinline__ v16h load_a16(const _Float16* row, int hf, int c) {
  return cat8(*(const v8h*)(row + c * 32 + hf * 16),
              *(const v8h*)(row + c * 32 + hf * 16 + 8));
}

// B-fragment (32x16 f16) from a padded LDS tile row pointer (16B aligned).
static __device__ __forceinline__ v16h load_b_lds(const _Float16* p) {
  return cat8(*(const v8h*)p, *(const v8h*)(p + 8));
}

// ---------------------------------------------------------------------------
// Tensor Data Mover: one instruction DMAs a 2D f16 tile (row padding per
// D#.pad_* so LDS rows land conflict-free). 5-arg builtin on ROCm7.2/clang-22,
// 6-arg on clang-23; cooperative-copy fallback if absent.
#if __has_builtin(__builtin_amdgcn_tensor_load_to_lds)
#define HAVE_TDM 1
typedef uint32_t u32x4 __attribute__((ext_vector_type(4)));
typedef int32_t  i32x4 __attribute__((ext_vector_type(4)));
typedef int32_t  i32x8 __attribute__((ext_vector_type(8)));

static __device__ __forceinline__ void tdm_load_tile_f16(
    uint32_t lds_addr, const _Float16* gptr,
    uint32_t tensor_d0, uint32_t tensor_d1,
    uint32_t tile_d0, uint32_t tile_d1,
    uint64_t row_stride_elems,
    uint32_t pad_interval_code, uint32_t pad_amount_code) {
  uint64_t ga = (uint64_t)(uintptr_t)gptr;
  u32x4 g0;
  g0.x = 1u;                                           // count=1, user descriptor
  g0.y = lds_addr;                                     // lds_addr [63:32]
  g0.z = (uint32_t)ga;                                 // global_addr [95:64]
  g0.w = (uint32_t)((ga >> 32) & 0x01FFFFFFu) | (2u << 30);  // addr hi + type=2
  i32x8 g1;
  // data_size=1 (2B), pad_enable, pad_interval, pad_amount
  g1[0] = (int)((1u << 16) | (1u << 20) | (pad_interval_code << 22) | (pad_amount_code << 25));
  g1[1] = (int)((tensor_d0 & 0xFFFFu) << 16);                        // tensor_dim0 lo
  g1[2] = (int)((tensor_d0 >> 16) | ((tensor_d1 & 0xFFFFu) << 16));  // dim0 hi | dim1 lo
  g1[3] = (int)((tensor_d1 >> 16) | (tile_d0 << 16));                // dim1 hi | tile_dim0
  g1[4] = (int)(tile_d1 & 0xFFFFu);                                  // tile_dim1 (tile_dim2=0)
  g1[5] = (int)(uint32_t)(row_stride_elems & 0xFFFFFFFFu);           // dim0_stride lo
  g1[6] = (int)(uint32_t)((row_stride_elems >> 32) & 0xFFFFu);       // dim0_stride hi
  g1[7] = 0;
  i32x4 z4 = {0, 0, 0, 0};
#if defined(__clang_major__) && __clang_major__ >= 23
  i32x8 z8 = {0, 0, 0, 0, 0, 0, 0, 0};
  __builtin_amdgcn_tensor_load_to_lds(g0, g1, z4, z4, z8, 0);
#else
  __builtin_amdgcn_tensor_load_to_lds(g0, g1, z4, z4, 0);
#endif
}
#else
#define HAVE_TDM 0
#endif

// ---------------------------------------------------------------------------
// Kernel 0a: Wq/Wk f32 [D,D] -> transposed f16 [out][in] (B-frag contiguity).
__global__ void cvt_transpose_w_kernel(const float* __restrict__ W, _Float16* __restrict__ WT) {
  int id = blockIdx.x * 256 + threadIdx.x;
  int o = id >> 8, i = id & 255;
  WT[(size_t)o * DIM + i] = (_Float16)W[(size_t)i * DIM + o];
}

// Kernel 0b: states f32 [N,D] -> f16 transposed [D,N].
__global__ void cvt_transpose_s_kernel(const float* __restrict__ S, _Float16* __restrict__ ST) {
  int id = blockIdx.x * 256 + threadIdx.x;
  int d = id >> 13, n = id & 8191;
  ST[(size_t)d * N_TOK + n] = (_Float16)S[(size_t)n * DIM + d];
}

// ---------------------------------------------------------------------------
// Kernel 1: Q = states@Wq, K = states@Wk (small: 2 GFLOP; direct global loads).
__global__ __launch_bounds__(256, 1)
void proj_kernel(const float* __restrict__ states,
                 const _Float16* __restrict__ WqT, const _Float16* __restrict__ WkT,
                 _Float16* __restrict__ Qh, _Float16* __restrict__ Kh) {
  const int lane = threadIdx.x & 31;
  const int wave = threadIdx.x >> 5;
  const int hf   = lane >> 4;
  const int ln   = lane & 15;
  const int r0   = (blockIdx.x * 8 + wave) * 16;

  v16h af[8];
  const float* srow = states + (size_t)(r0 + ln) * DIM;
#pragma unroll
  for (int c = 0; c < 8; ++c) {
    v8f lo = *(const v8f*)(srow + c * 32 + hf * 16);
    v8f hi = *(const v8f*)(srow + c * 32 + hf * 16 + 8);
    v16h a;
#pragma unroll
    for (int e = 0; e < 8; ++e) { a[e] = (_Float16)lo[e]; a[8 + e] = (_Float16)hi[e]; }
    af[c] = a;
  }

#pragma unroll
  for (int mat = 0; mat < 2; ++mat) {
    const _Float16* WT = mat ? WkT : WqT;
    _Float16* Out = mat ? Kh : Qh;
    for (int nc = 0; nc < 16; ++nc) {
      v8f acc = vzero8();
      const _Float16* wrow = WT + (size_t)(nc * 16 + ln) * DIM + hf * 16;
#pragma unroll
      for (int c = 0; c < 8; ++c)
        acc = wmma16(af[c], *(const v16h*)(wrow + c * 32), acc);
#pragma unroll
      for (int v = 0; v < 8; ++v)
        Out[(size_t)(r0 + hf * 8 + v) * DIM + nc * 16 + ln] = (_Float16)acc[v];
    }
  }
}

// ---------------------------------------------------------------------------
// Kernel 2: fused attention, partial over a j-range. Block = 8 waves * 16 rows.
// TDM double-buffers block-shared K tile [32][256+8] and statesT tile
// [256][32+8] in LDS; per-wave P tiles transit a private LDS buffer
// (C-layout -> A-layout). Softmax without max-subtraction (scores ~ N(0,1)).
__global__ __launch_bounds__(256, 1)
void attn_partial_kernel(const _Float16* __restrict__ Qh, const _Float16* __restrict__ Kh,
                         const _Float16* __restrict__ ShT,
                         float* __restrict__ Opart, float* __restrict__ rspart,
                         int jsplit, int jlen) {
  __shared__ __align__(16) _Float16 Kt[2][32][264];   // 528B row stride: b128 reads hit distinct bank groups
  __shared__ __align__(16) _Float16 St[2][256][40];   // 80B row stride: distinct bank groups
  __shared__ __align__(16) _Float16 ldsP[8][16][40];

  const int lane = threadIdx.x & 31;
  const int wave = threadIdx.x >> 5;
  const int hf   = lane >> 4;
  const int ln   = lane & 15;
  const int rb   = blockIdx.x / jsplit;
  const int js   = blockIdx.x % jsplit;
  const int r0   = (rb * 8 + wave) * 16;
  const int j0   = js * jlen;
  const float scale = 0.0625f;                         // 1/sqrt(256)

  v16h qf[8];
  const _Float16* qrow = Qh + (size_t)(r0 + ln) * DIM;
#pragma unroll
  for (int c = 0; c < 8; ++c) qf[c] = load_a16(qrow, hf, c);

  v8f o[16];
#pragma unroll
  for (int nc = 0; nc < 16; ++nc) o[nc] = vzero8();
  float rs[8] = {0.f,0.f,0.f,0.f,0.f,0.f,0.f,0.f};

  const int nIter = jlen / JCHUNK;
#if HAVE_TDM
  if (wave == 0) {
    tdm_load_tile_f16((uint32_t)(uintptr_t)&Kt[0][0][0], Kh + (size_t)j0 * DIM,
                      DIM, N_TOK, DIM, 32, DIM, 6u, 3u);       // pad 16B per 512B row
    tdm_load_tile_f16((uint32_t)(uintptr_t)&St[0][0][0], ShT + j0,
                      N_TOK, DIM, 32, DIM, N_TOK, 3u, 3u);     // pad 16B per 64B row
  }
#endif

  for (int it = 0; it < nIter; ++it) {
#if HAVE_TDM
    const int cur = it & 1;
    if (wave == 0) __builtin_amdgcn_s_wait_tensorcnt(0);
    __syncthreads();                                   // tile ready; prev reads of other buffer done
    if (wave == 0 && it + 1 < nIter) {
      const int jn = j0 + (it + 1) * JCHUNK;
      tdm_load_tile_f16((uint32_t)(uintptr_t)&Kt[cur ^ 1][0][0], Kh + (size_t)jn * DIM,
                        DIM, N_TOK, DIM, 32, DIM, 6u, 3u);
      tdm_load_tile_f16((uint32_t)(uintptr_t)&St[cur ^ 1][0][0], ShT + jn,
                        N_TOK, DIM, 32, DIM, N_TOK, 3u, 3u);
    }
#else
    const int cur = 0;
    const int j = j0 + it * JCHUNK;
    __syncthreads();                                   // protect prior reads
    for (int t = threadIdx.x; t < 1024; t += 256) {    // K tile: 32 rows x 32 chunks(16B)
      int r = t >> 5, c16 = t & 31;
      *(v8h*)&Kt[0][r][c16 * 8] = *(const v8h*)(Kh + (size_t)(j + r) * DIM + c16 * 8);
    }
    for (int t = threadIdx.x; t < 1024; t += 256) {    // ShT tile: 256 rows x 4 chunks
      int r = t >> 2, c16 = t & 3;
      *(v8h*)&St[0][r][c16 * 8] = *(const v8h*)(ShT + (size_t)r * N_TOK + j + c16 * 8);
    }
    __syncthreads();
#endif
    // ---- S tiles from LDS ----
    v8f s1 = vzero8(), s2 = vzero8();
    const _Float16* kp1 = &Kt[cur][ln][hf * 16];
    const _Float16* kp2 = &Kt[cur][16 + ln][hf * 16];
#pragma unroll
    for (int c = 0; c < 8; ++c) {
      s1 = wmma16(qf[c], load_b_lds(kp1 + c * 32), s1);
      s2 = wmma16(qf[c], load_b_lds(kp2 + c * 32), s2);
    }
    // ---- P = exp(S*scale), stash via private LDS for layout change ----
#pragma unroll
    for (int v = 0; v < 8; ++v) {
      float p1 = __expf(s1[v] * scale);
      float p2 = __expf(s2[v] * scale);
      rs[v] += p1 + p2;
      int m = hf * 8 + v;
      ldsP[wave][m][ln]      = (_Float16)p1;
      ldsP[wave][m][16 + ln] = (_Float16)p2;
    }
    v16h pa = cat8(*(const v8h*)&ldsP[wave][ln][hf * 16],
                   *(const v8h*)&ldsP[wave][ln][hf * 16 + 8]);
    // ---- O += P @ states[j:j+32, :] ----
#pragma unroll
    for (int nc = 0; nc < 16; ++nc)
      o[nc] = wmma16(pa, load_b_lds(&St[cur][nc * 16 + ln][hf * 16]), o[nc]);
  }

  // ---- write partial O and partial row sums (no normalization here) ----
#pragma unroll
  for (int nc = 0; nc < 16; ++nc)
#pragma unroll
    for (int v = 0; v < 8; ++v)
      Opart[((size_t)js * N_TOK + r0 + hf * 8 + v) * DIM + nc * 16 + ln] = o[nc][v];
#pragma unroll
  for (int v = 0; v < 8; ++v) {
    float t = rs[v];
    t += __shfl_xor(t, 1); t += __shfl_xor(t, 2);
    t += __shfl_xor(t, 4); t += __shfl_xor(t, 8);
    if (ln == 0) rspart[(size_t)js * N_TOK + r0 + hf * 8 + v] = t;
  }
}

// ---------------------------------------------------------------------------
// Kernel 2b: combine partials -> normalized attn (f16) + squared row norms.
__global__ __launch_bounds__(256)
void attn_combine_kernel(const float* __restrict__ Opart, const float* __restrict__ rspart,
                         _Float16* __restrict__ attnH, float* __restrict__ sq, int jsplit) {
  const int n = blockIdx.x;
  const int d = threadIdx.x;
  float rs = 0.f, acc = 0.f;
  for (int js = 0; js < jsplit; ++js) {
    rs  += rspart[(size_t)js * N_TOK + n];
    acc += Opart[((size_t)js * N_TOK + n) * DIM + d];
  }
  float val = acc / rs;
  attnH[(size_t)n * DIM + d] = (_Float16)val;
  __shared__ float red[256];
  red[d] = val * val;
  __syncthreads();
  for (int s = 128; s > 0; s >>= 1) {
    if (d < s) red[d] += red[d + s];
    __syncthreads();
  }
  if (d == 0) sq[n] = red[0];
}

// ---------------------------------------------------------------------------
// Kernel 3: fused RBF row-mean, partial over a j-range; TDM-staged attn tiles.
__global__ __launch_bounds__(256, 1)
void rbf_partial_kernel(const _Float16* __restrict__ attnH, const float* __restrict__ sq,
                        float* __restrict__ featpart, int jsplit, int jlen) {
  __shared__ __align__(16) _Float16 At[2][32][264];

  const int lane = threadIdx.x & 31;
  const int wave = threadIdx.x >> 5;
  const int hf   = lane >> 4;
  const int ln   = lane & 15;
  const int rb   = blockIdx.x / jsplit;
  const int js   = blockIdx.x % jsplit;
  const int r0   = (rb * 8 + wave) * 16;
  const int j0   = js * jlen;

  v16h af[8];
  const _Float16* arow = attnH + (size_t)(r0 + ln) * DIM;
#pragma unroll
  for (int c = 0; c < 8; ++c) af[c] = load_a16(arow, hf, c);

  float sqi[8];
#pragma unroll
  for (int v = 0; v < 8; ++v) sqi[v] = sq[r0 + hf * 8 + v];

  float acc[8] = {0.f,0.f,0.f,0.f,0.f,0.f,0.f,0.f};
  const int nIter = jlen / JCHUNK;
#if HAVE_TDM
  if (wave == 0)
    tdm_load_tile_f16((uint32_t)(uintptr_t)&At[0][0][0], attnH + (size_t)j0 * DIM,
                      DIM, N_TOK, DIM, 32, DIM, 6u, 3u);
#endif

  for (int it = 0; it < nIter; ++it) {
    const int j = j0 + it * JCHUNK;
#if HAVE_TDM
    const int cur = it & 1;
    if (wave == 0) __builtin_amdgcn_s_wait_tensorcnt(0);
    __syncthreads();
    if (wave == 0 && it + 1 < nIter)
      tdm_load_tile_f16((uint32_t)(uintptr_t)&At[cur ^ 1][0][0],
                        attnH + (size_t)(j + JCHUNK) * DIM, DIM, N_TOK, DIM, 32, DIM, 6u, 3u);
#else
    const int cur = 0;
    __syncthreads();
    for (int t = threadIdx.x; t < 1024; t += 256) {
      int r = t >> 5, c16 = t & 31;
      *(v8h*)&At[0][r][c16 * 8] = *(const v8h*)(attnH + (size_t)(j + r) * DIM + c16 * 8);
    }
    __syncthreads();
#endif
    v8f g1 = vzero8(), g2 = vzero8();
    const _Float16* p1 = &At[cur][ln][hf * 16];
    const _Float16* p2 = &At[cur][16 + ln][hf * 16];
#pragma unroll
    for (int c = 0; c < 8; ++c) {
      g1 = wmma16(af[c], load_b_lds(p1 + c * 32), g1);
      g2 = wmma16(af[c], load_b_lds(p2 + c * 32), g2);
    }
    float sqj1 = sq[j + ln];
    float sqj2 = sq[j + 16 + ln];
#pragma unroll
    for (int v = 0; v < 8; ++v) {
      float d1 = fmaxf(sqi[v] + sqj1 - 2.0f * g1[v], 0.0f);
      float d2 = fmaxf(sqi[v] + sqj2 - 2.0f * g2[v], 0.0f);
      acc[v] += __expf(-d1) + __expf(-d2);              // GAMMA = 1
    }
  }
#pragma unroll
  for (int v = 0; v < 8; ++v) {
    float t = acc[v];
    t += __shfl_xor(t, 1); t += __shfl_xor(t, 2);
    t += __shfl_xor(t, 4); t += __shfl_xor(t, 8);
    if (ln == 0) featpart[(size_t)js * N_TOK + r0 + hf * 8 + v] = t;
  }
}

// ---------------------------------------------------------------------------
// Kernel 4: feat = sum(featpart)/N, then MLP head.
__global__ void head_kernel(const float* __restrict__ featpart,
                            const float* __restrict__ W1, const float* __restrict__ b1,
                            const float* __restrict__ W2, const float* __restrict__ b2,
                            float* __restrict__ out, int jsplit) {
  int n = blockIdx.x * blockDim.x + threadIdx.x;
  if (n >= N_TOK) return;
  float f = 0.f;
  for (int js = 0; js < jsplit; ++js) f += featpart[(size_t)js * N_TOK + n];
  f *= (1.0f / (float)N_TOK);
  float acc = b2[0];
#pragma unroll
  for (int h = 0; h < 64; ++h) {
    float t = fmaxf(fmaf(f, W1[h], b1[h]), 0.0f);
    acc = fmaf(t, W2[h], acc);
  }
  out[n] = acc;
}

// ---------------------------------------------------------------------------
extern "C" void kernel_launch(void* const* d_in, const int* in_sizes, int n_in,
                              void* d_out, int out_size, void* d_ws, size_t ws_size,
                              hipStream_t stream) {
  (void)in_sizes; (void)n_in; (void)out_size;
  const float* states = (const float*)d_in[0];
  const float* Wq     = (const float*)d_in[1];
  const float* Wk     = (const float*)d_in[2];
  const float* W1     = (const float*)d_in[3];
  const float* b1     = (const float*)d_in[4];
  const float* W2     = (const float*)d_in[5];
  const float* b2     = (const float*)d_in[6];
  float* out = (float*)d_out;

  char* ws = (char*)d_ws;
  size_t off = 0;
  auto carve = [&](size_t bytes) { char* p = ws + off; off += (bytes + 255) & ~(size_t)255; return p; };
  _Float16* Qh    = (_Float16*)carve((size_t)N_TOK * DIM * 2);
  _Float16* Kh    = (_Float16*)carve((size_t)N_TOK * DIM * 2);
  _Float16* attnH = (_Float16*)carve((size_t)N_TOK * DIM * 2);
  _Float16* ShT   = (_Float16*)carve((size_t)DIM * N_TOK * 2);
  _Float16* WqT   = (_Float16*)carve((size_t)DIM * DIM * 2);
  _Float16* WkT   = (_Float16*)carve((size_t)DIM * DIM * 2);
  float*    sqv   = (float*)carve((size_t)N_TOK * 4);

  // Split the j-reduction up to 4 ways for occupancy, if workspace allows.
  const size_t per_split = (size_t)N_TOK * DIM * 4 + 2 * (size_t)N_TOK * 4 + 3 * 256;
  int jsplit = 4;
  while (jsplit > 1 && off + (size_t)jsplit * per_split > ws_size) jsplit >>= 1;
  float* Opart    = (float*)carve((size_t)jsplit * N_TOK * DIM * 4);
  float* rspart   = (float*)carve((size_t)jsplit * N_TOK * 4);
  float* featpart = (float*)carve((size_t)jsplit * N_TOK * 4);
  const int jlen = N_TOK / jsplit;

  cvt_transpose_w_kernel<<<(DIM * DIM) / 256, 256, 0, stream>>>(Wq, WqT);
  cvt_transpose_w_kernel<<<(DIM * DIM) / 256, 256, 0, stream>>>(Wk, WkT);
  cvt_transpose_s_kernel<<<(DIM * N_TOK) / 256, 256, 0, stream>>>(states, ShT);

  const int rowBlocks = N_TOK / (16 * 8);              // 64
  proj_kernel<<<rowBlocks, 256, 0, stream>>>(states, WqT, WkT, Qh, Kh);
  attn_partial_kernel<<<rowBlocks * jsplit, 256, 0, stream>>>(Qh, Kh, ShT, Opart, rspart, jsplit, jlen);
  attn_combine_kernel<<<N_TOK, 256, 0, stream>>>(Opart, rspart, attnH, sqv, jsplit);
  rbf_partial_kernel<<<rowBlocks * jsplit, 256, 0, stream>>>(attnH, sqv, featpart, jsplit, jlen);
  head_kernel<<<N_TOK / 256, 256, 0, stream>>>(featpart, W1, b1, W2, b2, out, jsplit);
}
